// Mamba2Block_63677185130931
// MI455X (gfx1250) — matrix operand order repaired and checked by reading, not compiled, not run on hardware
//
#include <hip/hip_runtime.h>
#include <hip/hip_bf16.h>

// ---------------------------------------------------------------------------
// Mamba2 block for gfx1250 (MI455X).
//   HID=2048, STATE=128, HEADS=64, HEADDIM=64, GROUPS=1, K=4,
//   INTER=4096, CONV_DIM=4352, PROJ=8512, b=2, s=2048 -> ROWS=4096
// ---------------------------------------------------------------------------

#define HID      2048
#define STATE    128
#define HEADS    64
#define HEADDIM  64
#define INTER    4096
#define CONVD    4352
#define PROJ     8512
#define ROWS     4096
#define SEQ      2048
#define EPS      1e-5f

// Flip to 0 if the async-to-LDS inline asm fails to assemble.
#define USE_ASYNC_LDS 1

typedef __attribute__((ext_vector_type(16))) __bf16 v16bf;
typedef __attribute__((ext_vector_type(8)))  float  v8f;

// ---------------------------------------------------------------------------
// CDNA5 async global->LDS helpers (ASYNCcnt-tracked DMA, ISA ch.10 async ops)
// ---------------------------------------------------------------------------
__device__ __forceinline__ void async_copy_b128(unsigned lds_byte_off,
                                                const void* gaddr) {
#if USE_ASYNC_LDS
  asm volatile("global_load_async_to_lds_b128 %0, %1, off"
               :: "v"(lds_byte_off), "v"(gaddr) : "memory");
#else
  (void)lds_byte_off; (void)gaddr;
#endif
}
__device__ __forceinline__ void wait_async_all() {
#if USE_ASYNC_LDS
  asm volatile("s_wait_asynccnt 0x0" ::: "memory");
#endif
}

// ---------------------------------------------------------------------------
// f32 -> bf16 elementwise conversion (weights)
// ---------------------------------------------------------------------------
__global__ __launch_bounds__(256) void f32_to_bf16_kernel(
    const float* __restrict__ in, __bf16* __restrict__ out, int n) {
  int i = blockIdx.x * 256 + threadIdx.x;
  if (i < n) out[i] = (__bf16)in[i];
}

// ---------------------------------------------------------------------------
// RMSNorm over HID, writes bf16 activations for GEMM1.
// ---------------------------------------------------------------------------
__global__ __launch_bounds__(256) void rmsnorm_bf16_kernel(
    const float* __restrict__ x, const float* __restrict__ w,
    __bf16* __restrict__ out) {
  const int row = blockIdx.x;
  const int tid = threadIdx.x;
  float vals[8];
  float partial = 0.f;
#pragma unroll
  for (int u = 0; u < 8; ++u) {
    int c = u * 256 + tid;
    float v = x[(size_t)row * HID + c];
    vals[u] = v;
    partial += v * v;
  }
  __shared__ float red[256];
  red[tid] = partial;
  __syncthreads();
  for (int s = 128; s > 0; s >>= 1) {
    if (tid < s) red[tid] += red[tid + s];
    __syncthreads();
  }
  float rs = rsqrtf(red[0] * (1.0f / HID) + EPS);
#pragma unroll
  for (int u = 0; u < 8; ++u) {
    int c = u * 256 + tid;
    out[(size_t)row * HID + c] = (__bf16)(vals[u] * rs * w[c]);
  }
}

// ---------------------------------------------------------------------------
// bf16 WMMA GEMM:  C[M,N] = A[M,K] * B[N,K]^T (+ optional f32 addend)
// Block tile 256x128, 8 waves laid out 4(M) x 2(N); each wave owns a
// 64x64 sub-tile = 4x4 v_wmma_f32_16x16x32_bf16 accumulators.
// Tiles staged into double-buffered LDS via async global->LDS DMA; one
// s_wait_asynccnt + one barrier per K=32 chunk; next-chunk DMA overlaps
// the current chunk's wmma stream.
// ---------------------------------------------------------------------------
#define KC   32
#define LDA  40                  // bf16 elems per LDS row (80 B, 16B aligned)
#define SA_E (256 * LDA)         // A buffer elems (20480 B)
#define SB_E (128 * LDA)         // B buffer elems (10240 B)

__global__ __launch_bounds__(256) void gemm_bf16_wmma_kernel(
    const __bf16* __restrict__ A, const __bf16* __restrict__ B,
    const float* __restrict__ addend, float* __restrict__ C,
    int M, int N, int K) {
  __shared__ __bf16 sA[2 * SA_E];
  __shared__ __bf16 sB[2 * SB_E];

  const int tid  = threadIdx.x;
  const int lane = tid & 31;
  const int wave = tid >> 5;        // 0..7
  const int waveM = wave >> 1;      // 0..3 -> 64-row band
  const int waveN = wave & 1;       // 0..1 -> 64-col band

  const int blockColBase = blockIdx.x * 128;
  const int blockRowBase = blockIdx.y * 256;

  const unsigned sAoff = (unsigned)(size_t)(void*)sA;
  const unsigned sBoff = (unsigned)(size_t)(void*)sB;

  const v8f vzero = {0.f, 0.f, 0.f, 0.f, 0.f, 0.f, 0.f, 0.f};
  v8f acc[4][4];
#pragma unroll
  for (int mi = 0; mi < 4; ++mi)
#pragma unroll
    for (int ni = 0; ni < 4; ++ni) acc[mi][ni] = vzero;

  // --- cooperative tile DMA: A row per thread (64B), B half-row (32B) ---
  const int brow = tid >> 1;
  const int bc16 = (tid & 1) * 16;            // bf16 col offset in chunk
  const int bInBounds = (blockColBase + brow) < N;

  auto load_tiles = [&](int k0, int buf) {
#if USE_ASYNC_LDS
    const __bf16* ga = A + (size_t)(blockRowBase + tid) * K + k0;
    unsigned la = sAoff + (unsigned)(buf * SA_E + tid * LDA) * 2u;
    async_copy_b128(la,      ga);
    async_copy_b128(la + 16, ga + 8);
    async_copy_b128(la + 32, ga + 16);
    async_copy_b128(la + 48, ga + 24);
    if (bInBounds) {
      const __bf16* gb =
          B + (size_t)(blockColBase + brow) * K + k0 + bc16;
      unsigned lb = sBoff + (unsigned)(buf * SB_E + brow * LDA + bc16) * 2u;
      async_copy_b128(lb,      gb);
      async_copy_b128(lb + 16, gb + 8);
    }
#else
    const __bf16* ga = A + (size_t)(blockRowBase + tid) * K + k0;
    __bf16* la = &sA[buf * SA_E + tid * LDA];
    *(uint4*)(la)      = *(const uint4*)(ga);
    *(uint4*)(la + 8)  = *(const uint4*)(ga + 8);
    *(uint4*)(la + 16) = *(const uint4*)(ga + 16);
    *(uint4*)(la + 24) = *(const uint4*)(ga + 24);
    if (bInBounds) {
      const __bf16* gb =
          B + (size_t)(blockColBase + brow) * K + k0 + bc16;
      __bf16* lb = &sB[buf * SB_E + brow * LDA + bc16];
      *(uint4*)(lb)     = *(const uint4*)(gb);
      *(uint4*)(lb + 8) = *(const uint4*)(gb + 8);
    }
#endif
  };

  load_tiles(0, 0);
  int buf = 0;

  for (int k0 = 0; k0 < K; k0 += KC) {
    wait_async_all();    // this wave's DMA for chunk k0 has landed in LDS
    __syncthreads();     // every wave's DMA landed; prev-buf readers done

    if (k0 + KC < K) load_tiles(k0 + KC, buf ^ 1);   // overlap next DMA

    // ---- load ALL fragments first (one dscnt wait), then 16 wmmas ----
    union frag { v16bf v; uint4 q[2]; };
    frag af[4], bfr[4];
    const int akh = (lane >> 4) * 8;    // A: K offset per lane half
    const int bkh = (lane >> 4) * 8;    // B: K offset per lane half
    const __bf16* sAb = &sA[buf * SA_E];
    const __bf16* sBb = &sB[buf * SB_E];
#pragma unroll
    for (int mi = 0; mi < 4; ++mi) {
      int r = waveM * 64 + mi * 16 + (lane & 15);
      af[mi].q[0] = *(const uint4*)&sAb[r * LDA + akh];
      af[mi].q[1] = *(const uint4*)&sAb[r * LDA + 16 + akh];
    }
#pragma unroll
    for (int ni = 0; ni < 4; ++ni) {
      int c = waveN * 64 + ni * 16 + (lane & 15);
      bfr[ni].q[0] = *(const uint4*)&sBb[c * LDA + bkh];
      bfr[ni].q[1] = *(const uint4*)&sBb[c * LDA + 16 + bkh];
    }
#pragma unroll
    for (int mi = 0; mi < 4; ++mi)
#pragma unroll
      for (int ni = 0; ni < 4; ++ni)
        acc[mi][ni] = __builtin_amdgcn_wmma_f32_16x16x32_bf16(
            false, af[mi].v, false, bfr[ni].v, (short)0, acc[mi][ni],
            false, false);
    buf ^= 1;
  }

  // Epilogue: C/D layout -> row = (lane>>4)*8 + j, col = lane&15
  const int lrow = (lane >> 4) * 8;
  const int lcol = lane & 15;
#pragma unroll
  for (int mi = 0; mi < 4; ++mi) {
#pragma unroll
    for (int ni = 0; ni < 4; ++ni) {
      int colBase = blockColBase + waveN * 64 + ni * 16 + lcol;
      if (colBase < N) {
#pragma unroll
        for (int j = 0; j < 8; ++j) {
          int row = blockRowBase + waveM * 64 + mi * 16 + lrow + j;
          size_t idx = (size_t)row * N + colBase;
          float v = acc[mi][ni][j];
          if (addend) v += addend[idx];
          C[idx] = v;
        }
      }
    }
  }
}

// ---------------------------------------------------------------------------
// dt = softplus(proj_dt + dt_bias); dA = exp(dt * (-exp(A_log)))
// ---------------------------------------------------------------------------
__global__ __launch_bounds__(256) void dt_dA_kernel(
    const float* __restrict__ proj, const float* __restrict__ dt_bias,
    const float* __restrict__ A_log, float* __restrict__ dtv,
    float* __restrict__ dAv) {
  int idx = blockIdx.x * 256 + threadIdx.x;
  if (idx >= ROWS * HEADS) return;
  int h = idx & (HEADS - 1);
  int row = idx >> 6;
  float x = proj[(size_t)row * PROJ + (INTER + CONVD) + h] + dt_bias[h];
  float sp = (x > 20.f) ? x : log1pf(__expf(x));
  float a = -__expf(A_log[h]);
  dtv[idx] = sp;
  dAv[idx] = __expf(sp * a);
}

// ---------------------------------------------------------------------------
// Causal depthwise conv1d (K=4) + bias + SiLU.
// ---------------------------------------------------------------------------
__global__ __launch_bounds__(256) void conv_silu_kernel(
    const float* __restrict__ proj, const float* __restrict__ cw,
    const float* __restrict__ cb, float* __restrict__ xbca) {
  const int c  = blockIdx.x * 256 + threadIdx.x;
  const int t0 = blockIdx.y * 128;
  const int bb = blockIdx.z;
  const float w0 = cw[c * 4 + 0], w1 = cw[c * 4 + 1];
  const float w2 = cw[c * 4 + 2], w3 = cw[c * 4 + 3];
  const float bias = cb[c];

  float x0 = 0.f, x1 = 0.f, x2 = 0.f;
#pragma unroll
  for (int j = 0; j < 3; ++j) {
    int tj = t0 - 3 + j;
    float v = (tj >= 0)
        ? proj[(size_t)(bb * SEQ + tj) * PROJ + INTER + c] : 0.f;
    if (j == 0) x0 = v; else if (j == 1) x1 = v; else x2 = v;
  }
  for (int i = 0; i < 128; ++i) {
    int t = t0 + i;
    float x3 = proj[(size_t)(bb * SEQ + t) * PROJ + INTER + c];
    float v = x0 * w0 + x1 * w1 + x2 * w2 + x3 * w3 + bias;
    v = v / (1.f + __expf(-v));
    xbca[(size_t)(bb * SEQ + t) * CONVD + c] = v;
    x0 = x1; x1 = x2; x2 = x3;
  }
}

// ---------------------------------------------------------------------------
// Selective scan.  One block per (head, batch); 256 threads.
// Thread (p=tid>>2, q=tid&3) owns state[p][q*32..q*32+31] in registers.
// ---------------------------------------------------------------------------
#define TCHUNK 32

__global__ __launch_bounds__(256) void scan_kernel(
    const float* __restrict__ xbca, const float* __restrict__ dtv,
    const float* __restrict__ dAv, const float* __restrict__ Dparm,
    float* __restrict__ y) {
  const int h  = blockIdx.x;
  const int bb = blockIdx.y;
  const int tid = threadIdx.x;
  const int p = tid >> 2;
  const int q = tid & 3;
  const float Dh = Dparm[h];

  __shared__ float sB[TCHUNK][STATE];
  __shared__ float sC[TCHUNK][STATE];
  __shared__ float sX[TCHUNK][HEADDIM];
  __shared__ float sdA[TCHUNK];
  __shared__ float sdt[TCHUNK];

  float st[32];
#pragma unroll
  for (int j = 0; j < 32; ++j) st[j] = 0.f;

  const int ltt = tid >> 3;
  const int lb  = (tid & 7) * 16;
  const int lx  = (tid & 7) * 8;

  for (int t0 = 0; t0 < SEQ; t0 += TCHUNK) {
    size_t rowAddr = (size_t)(bb * SEQ + t0 + ltt) * CONVD;
#pragma unroll
    for (int u = 0; u < 4; ++u) {
      *(float4*)&sB[ltt][lb + u * 4] =
          *(const float4*)&xbca[rowAddr + INTER + lb + u * 4];
      *(float4*)&sC[ltt][lb + u * 4] =
          *(const float4*)&xbca[rowAddr + INTER + STATE + lb + u * 4];
    }
    *(float4*)&sX[ltt][lx] =
        *(const float4*)&xbca[rowAddr + h * HEADDIM + lx];
    *(float4*)&sX[ltt][lx + 4] =
        *(const float4*)&xbca[rowAddr + h * HEADDIM + lx + 4];
    if (tid < TCHUNK) {
      int r = (bb * SEQ + t0 + tid) * HEADS + h;
      sdA[tid] = dAv[r];
      sdt[tid] = dtv[r];
    }
    __syncthreads();

    for (int tt = 0; tt < TCHUNK; ++tt) {
      float dAt = sdA[tt];
      float xp  = sX[tt][p];
      float dtx = sdt[tt] * xp;
      const float* Bp = &sB[tt][q * 32];
      const float* Cp = &sC[tt][q * 32];
      float ys = 0.f;
#pragma unroll
      for (int j = 0; j < 32; ++j) {
        st[j] = st[j] * dAt + dtx * Bp[j];
        ys += st[j] * Cp[j];
      }
      ys += __shfl_down(ys, 2, 4);
      ys += __shfl_down(ys, 1, 4);
      if (q == 0) {
        y[(size_t)(bb * SEQ + t0 + tt) * INTER + h * HEADDIM + p] =
            ys + Dh * xp;
      }
    }
    __syncthreads();
  }
}

// ---------------------------------------------------------------------------
// Gated RMSNorm: Yn = rmsnorm(y * silu(gate)) * gw  -> bf16
// ---------------------------------------------------------------------------
__global__ __launch_bounds__(256) void gated_norm_bf16_kernel(
    const float* __restrict__ y, const float* __restrict__ proj,
    const float* __restrict__ gw, __bf16* __restrict__ out) {
  const int row = blockIdx.x;
  const int tid = threadIdx.x;
  float tv[16];
  float partial = 0.f;
#pragma unroll
  for (int u = 0; u < 16; ++u) {
    int c = u * 256 + tid;
    float g  = proj[(size_t)row * PROJ + c];
    float yv = y[(size_t)row * INTER + c];
    float t  = yv * (g / (1.f + __expf(-g)));
    tv[u] = t;
    partial += t * t;
  }
  __shared__ float red[256];
  red[tid] = partial;
  __syncthreads();
  for (int s = 128; s > 0; s >>= 1) {
    if (tid < s) red[tid] += red[tid + s];
    __syncthreads();
  }
  float rs = rsqrtf(red[0] * (1.0f / INTER) + EPS);
#pragma unroll
  for (int u = 0; u < 16; ++u) {
    int c = u * 256 + tid;
    out[(size_t)row * INTER + c] = (__bf16)(tv[u] * rs * gw[c]);
  }
}

// ---------------------------------------------------------------------------
// Host-side orchestration
// ---------------------------------------------------------------------------
static inline size_t align256(size_t v) { return (v + 255) & ~(size_t)255; }

extern "C" void kernel_launch(void* const* d_in, const int* in_sizes, int n_in,
                              void* d_out, int out_size, void* d_ws,
                              size_t ws_size, hipStream_t stream) {
  (void)in_sizes; (void)n_in; (void)out_size; (void)ws_size;
  const float* hidden     = (const float*)d_in[0];
  const float* norm_w     = (const float*)d_in[1];
  const float* in_proj_w  = (const float*)d_in[2];
  const float* conv_w     = (const float*)d_in[3];
  const float* conv_b     = (const float*)d_in[4];
  const float* dt_bias    = (const float*)d_in[5];
  const float* A_log      = (const float*)d_in[6];
  const float* Dparm      = (const float*)d_in[7];
  const float* gate_nw    = (const float*)d_in[8];
  const float* out_proj_w = (const float*)d_in[9];
  float* out = (float*)d_out;

  char* ws = (char*)d_ws;
  size_t off = 0;
  auto carve = [&](size_t bytes) -> void* {
    void* p = ws + off;
    off = align256(off + bytes);
    return p;
  };
  __bf16* Xn   = (__bf16*)carve((size_t)ROWS * HID * 2);
  __bf16* W1b  = (__bf16*)carve((size_t)PROJ * HID * 2);
  __bf16* W2b  = (__bf16*)carve((size_t)HID * INTER * 2);
  float*  proj = (float*) carve((size_t)ROWS * PROJ * 4);
  float*  xbca = (float*) carve((size_t)ROWS * CONVD * 4);
  float*  dtv  = (float*) carve((size_t)ROWS * HEADS * 4);
  float*  dAv  = (float*) carve((size_t)ROWS * HEADS * 4);
  float*  ybuf = (float*) carve((size_t)ROWS * INTER * 4);
  __bf16* Yn   = (__bf16*)carve((size_t)ROWS * INTER * 2);

  {
    int n1 = PROJ * HID;
    f32_to_bf16_kernel<<<(n1 + 255) / 256, 256, 0, stream>>>(in_proj_w, W1b, n1);
    int n2 = HID * INTER;
    f32_to_bf16_kernel<<<(n2 + 255) / 256, 256, 0, stream>>>(out_proj_w, W2b, n2);
  }
  rmsnorm_bf16_kernel<<<ROWS, 256, 0, stream>>>(hidden, norm_w, Xn);
  {
    dim3 grid((PROJ + 127) / 128, ROWS / 256);
    gemm_bf16_wmma_kernel<<<grid, 256, 0, stream>>>(
        Xn, W1b, (const float*)nullptr, proj, ROWS, PROJ, HID);
  }
  dt_dA_kernel<<<(ROWS * HEADS + 255) / 256, 256, 0, stream>>>(
      proj, dt_bias, A_log, dtv, dAv);
  {
    dim3 grid(CONVD / 256, SEQ / 128, 2);
    conv_silu_kernel<<<grid, 256, 0, stream>>>(proj, conv_w, conv_b, xbca);
  }
  {
    dim3 grid(HEADS, 2);
    scan_kernel<<<grid, 256, 0, stream>>>(xbca, dtv, dAv, Dparm, ybuf);
  }
  gated_norm_bf16_kernel<<<ROWS, 256, 0, stream>>>(ybuf, proj, gate_nw, Yn);
  {
    dim3 grid(HID / 128, ROWS / 256);
    gemm_bf16_wmma_kernel<<<grid, 256, 0, stream>>>(
        Yn, W2b, hidden, out, ROWS, HID, INTER);
  }
}
